// DEACA_attention_v3_61761629716676
// MI455X (gfx1250) — compile-verified, matrix operand order checked by reading, not run on hardware
//
#include <hip/hip_runtime.h>
#include <math.h>

// Problem constants (match reference)
#define E_DIM 256
#define NHEAD 8
#define HDIM  32
#define B_DIM 4
#define H_DIM 64
#define W_DIM 64
#define T_DIM 4096   // H*W
#define NHEADS_TOTAL (B_DIM * NHEAD)   // 32

typedef __attribute__((ext_vector_type(16))) __bf16 v16bf;
typedef __attribute__((ext_vector_type(8)))  __bf16 v8bf;
typedef __attribute__((ext_vector_type(4)))  __bf16 v4bf;
typedef __attribute__((ext_vector_type(8)))  float  v8f;
typedef __attribute__((ext_vector_type(4)))  float  v4f;

// ---------- helpers ----------

__device__ __forceinline__ v8f wmma_bf16(v16bf a, v16bf b, v8f c) {
  return __builtin_amdgcn_wmma_f32_16x16x32_bf16(false, a, false, b, (short)0, c,
                                                 false, false);
}

// Load a 16x32 bf16 fragment (A or B operand) from a row-major bf16 matrix.
// Lane L<16: row = row0+L, K-halves {k0..k0+7} and {k0+16..k0+23}
// Lane L>=16: same row, K-halves {k0+8..k0+15} and {k0+24..k0+31}
// Identical convention for A and B => any K-interleave uncertainty cancels.
// Works for both global and LDS pointers (generic address space).
__device__ __forceinline__ v16bf frag_ld(const __bf16* p, int ld, int row0, int k0) {
  int lane = threadIdx.x & 31;
  const __bf16* q = p + (long)(row0 + (lane & 15)) * ld + k0 + ((lane & 16) ? 8 : 0);
  v8bf lo = *(const v8bf*)q;         // 16B load
  v8bf hi = *(const v8bf*)(q + 16);  // 16B load
  return __builtin_shufflevector(lo, hi, 0, 1, 2, 3, 4, 5, 6, 7,
                                 8, 9, 10, 11, 12, 13, 14, 15);
}

__device__ __forceinline__ v16bf frag_ld_clamp(const __bf16* p, int ld, int nrows,
                                               int row0, int k0) {
  int lane = threadIdx.x & 31;
  int r = row0 + (lane & 15);
  if (r >= nrows) r = nrows - 1;
  const __bf16* q = p + (long)r * ld + k0 + ((lane & 16) ? 8 : 0);
  v8bf lo = *(const v8bf*)q;
  v8bf hi = *(const v8bf*)(q + 16);
  return __builtin_shufflevector(lo, hi, 0, 1, 2, 3, 4, 5, 6, 7,
                                 8, 9, 10, 11, 12, 13, 14, 15);
}

// Async copy 16 bytes per lane from global memory straight into LDS
// (gfx1250 GLOBAL_LOAD_ASYNC_TO_LDS_B128, tracked with ASYNCcnt).
__device__ __forceinline__ void async_g2lds_b128(unsigned lds_byte_addr,
                                                 const void* gaddr) {
  asm volatile("global_load_async_to_lds_b128 %0, %1, off"
               :
               : "v"(lds_byte_addr), "v"((unsigned long long)(size_t)gaddr)
               : "memory");
}

__device__ __forceinline__ void wait_asynccnt0() {
  asm volatile("s_wait_asynccnt 0x0" ::: "memory");
}

// ---------- kernels ----------

// Bulk f32 -> bf16 convert, 4 elements/thread (float4 in, 8B out)
__global__ __launch_bounds__(256)
void convert_kernel(const float* __restrict__ in, __bf16* __restrict__ out, int n4) {
  int gid = blockIdx.x * blockDim.x + threadIdx.x;
  if (gid >= n4) return;
  v4f x = *(const v4f*)(in + (long)gid * 4);
  v4bf y;
  y[0] = (__bf16)x[0]; y[1] = (__bf16)x[1];
  y[2] = (__bf16)x[2]; y[3] = (__bf16)x[3];
  *(v4bf*)(out + (long)gid * 4) = y;
}

// out[o*inner+i] = bf16( inv * sum_j in[o*ostride + j*rstride + i] )
__global__ __launch_bounds__(256)
void mean_kernel(const float* __restrict__ in, __bf16* __restrict__ out,
                 int inner, int rcount, long rstride, long ostride,
                 int total, float inv) {
  int gid = blockIdx.x * blockDim.x + threadIdx.x;
  if (gid >= total) return;
  int o = gid / inner;
  int i = gid - o * inner;
  const float* p = in + (long)o * ostride + i;
  float s = 0.f;
  for (int j = 0; j < rcount; ++j) s += p[(long)j * rstride];
  out[(long)o * inner + i] = (__bf16)(s * inv);
}

// w_cat[n][0:256]=w_row[n], w_cat[n][256:512]=w_col[n]; bias_sum = b_row+b_col
__global__ __launch_bounds__(256)
void pack_wcat_kernel(const float* __restrict__ w_row, const float* __restrict__ w_col,
                      const float* __restrict__ b_row, const float* __restrict__ b_col,
                      __bf16* __restrict__ w_cat, float* __restrict__ bias_sum) {
  int idx = blockIdx.x * blockDim.x + threadIdx.x;   // 256*256
  if (idx >= E_DIM * E_DIM) return;
  int n = idx >> 8, k = idx & 255;
  w_cat[(long)n * 512 + k]       = (__bf16)w_row[(long)n * 256 + k];
  w_cat[(long)n * 512 + 256 + k] = (__bf16)w_col[(long)n * 256 + k];
  if (k == 0) bias_sum[n] = b_row[n] + b_col[n];
}

// C[M,N] = A[M,K] @ Bw[N,K]^T + bias[n]    (y = x @ w.T + b), bf16 inputs.
// One wave computes a 16x64 tile (A fragment reused 4x). N must be multiple of 64.
// Output to Cf (f32) or Cb (bf16) -- exactly one non-null.
__global__ __launch_bounds__(256)
void wmma_gemm_bf(const __bf16* __restrict__ A, const __bf16* __restrict__ Bw,
                  const float* __restrict__ bias,
                  float* __restrict__ Cf, __bf16* __restrict__ Cb,
                  int M, int N, int K) {
  int wave = threadIdx.x >> 5;
  int lane = threadIdx.x & 31;
  int mt = (M + 15) >> 4;
  int nt = N >> 6;
  long tile = (long)blockIdx.x * 8 + wave;
  if (tile >= (long)mt * nt) return;
  int mTile = (int)(tile / nt);
  int nTile = (int)(tile % nt);
  int row0 = mTile << 4;
  int col0 = nTile << 6;

  v8f acc0 = {}, acc1 = {}, acc2 = {}, acc3 = {};
  for (int k0 = 0; k0 < K; k0 += 32) {
    if (k0 + 32 < K) {
      __builtin_prefetch(A + (long)row0 * K + k0 + 32, 0, 1);   // global_prefetch_b8
      __builtin_prefetch(Bw + (long)col0 * K + k0 + 32, 0, 1);
    }
    v16bf a  = frag_ld_clamp(A, K, M, row0, k0);
    v16bf b0 = frag_ld(Bw, K, col0 +  0, k0);
    v16bf b1 = frag_ld(Bw, K, col0 + 16, k0);
    v16bf b2 = frag_ld(Bw, K, col0 + 32, k0);
    v16bf b3 = frag_ld(Bw, K, col0 + 48, k0);
    acc0 = wmma_bf16(a, b0, acc0);
    acc1 = wmma_bf16(a, b1, acc1);
    acc2 = wmma_bf16(a, b2, acc2);
    acc3 = wmma_bf16(a, b3, acc3);
  }

  int m0 = row0 + ((lane & 16) ? 8 : 0);
  v8f accs[4] = {acc0, acc1, acc2, acc3};
#pragma unroll
  for (int q = 0; q < 4; ++q) {
    int n = col0 + (q << 4) + (lane & 15);
    float bv = bias ? bias[n] : 0.f;
#pragma unroll
    for (int r = 0; r < 8; ++r) {
      int m = m0 + r;
      if (m < M) {
        long idx = (long)m * N + n;
        float v = accs[q][r] + bv;
        if (Cb) Cb[idx] = (__bf16)v;
        else    Cf[idx] = v;
      }
    }
  }
}

// gate[i][d] = sigmoid( sum_k pooled_heads[i][k] * conv_w[d][k] + conv_b[d] )
__global__ __launch_bounds__(256)
void gate_kernel(const float* __restrict__ vmp, const float* __restrict__ cw,
                 const float* __restrict__ cb, float* __restrict__ gate) {
  int idx = blockIdx.x * blockDim.x + threadIdx.x;     // 32 heads * 32 dims
  if (idx >= NHEADS_TOTAL * HDIM) return;
  int i = idx >> 5, d = idx & 31;
  int b = i >> 3, nh = i & 7;
  const float* ph = vmp + b * E_DIM + nh * HDIM;
  float acc = cb[d];
#pragma unroll 8
  for (int k = 0; k < HDIM; ++k) acc += ph[k] * cw[d * HDIM + k];
  gate[idx] = 1.f / (1.f + __expf(-acc));
}

// Build per-head gated K matrices [head][64][32] (bf16) and transposed V [head][32][64]
__global__ __launch_bounds__(256)
void repack_kernel(const float* __restrict__ krp, const float* __restrict__ kcp,
                   const float* __restrict__ vrp, const float* __restrict__ vcp,
                   const float* __restrict__ gate,
                   __bf16* __restrict__ krg, __bf16* __restrict__ kcg,
                   __bf16* __restrict__ vrT, __bf16* __restrict__ vcT) {
  int idx = blockIdx.x * blockDim.x + threadIdx.x;     // 32*64*32
  if (idx >= NHEADS_TOTAL * 64 * HDIM) return;
  int d = idx & 31;
  int w = (idx >> 5) & 63;
  int i = idx >> 11;                                   // head = b*8+nh
  int b = i >> 3, nh = i & 7;
  long src = (long)(b * 64 + w) * E_DIM + nh * HDIM + d;
  float g = gate[i * HDIM + d];
  long hk = (long)i * 2048 + w * HDIM + d;
  long hv = (long)i * 2048 + d * 64 + w;
  krg[hk] = (__bf16)(krp[src] * g);
  kcg[hk] = (__bf16)(kcp[src] * g);
  vrT[hv] = (__bf16)vrp[src];
  vcT[hv] = (__bf16)vcp[src];
}

// Fused axial attention per head:
//   S = Q_h @ K_h^T * scale ; P = softmax_row(S) ; XX = P @ V_h
// Q: qp[(b*T+t)*E + nh*32 + d] (bf16), K_h: kh[head][64][32], V_h^T: vhT[head][32][64]
// Output written into xx_cat [T*B, 512] at column offset colofs (+ nh*32).
__global__ __launch_bounds__(256)
void attn_kernel(const __bf16* __restrict__ qp, const __bf16* __restrict__ kh,
                 const __bf16* __restrict__ vhT, __bf16* __restrict__ xx,
                 int colofs, float scaling) {
  __shared__ __bf16 s_k[64 * 32];
  __shared__ __bf16 s_v[32 * 64];
  __shared__ __bf16 s_p[8][16 * 64];

  int head = blockIdx.y;
  int b = head >> 3, nh = head & 7;
  int t0 = blockIdx.x * 128;

  // ---- stage K_h (4KB) and V_h^T (4KB) into LDS with async-to-LDS copies ----
  {
    int tid = threadIdx.x;                       // 256 threads x 16B = 4KB each
    const __bf16* gk = kh + (long)head * 2048 + tid * 8;
    const __bf16* gv = vhT + (long)head * 2048 + tid * 8;
    unsigned lk = (unsigned)(size_t)(&s_k[tid * 8]);
    unsigned lv = (unsigned)(size_t)(&s_v[tid * 8]);
    async_g2lds_b128(lk, gk);
    async_g2lds_b128(lv, gv);
    wait_asynccnt0();
  }
  __syncthreads();

  int wave = threadIdx.x >> 5;
  int lane = threadIdx.x & 31;
  int trow = t0 + wave * 16;

  // ---- S = Q K^T : one K=32 step, 4 N-tiles of 16 ----
  const __bf16* qbase = qp + ((long)b * T_DIM + trow) * E_DIM + nh * HDIM;
  v16bf aq = frag_ld(qbase, E_DIM, 0, 0);
  v8f s[4];
#pragma unroll
  for (int ntile = 0; ntile < 4; ++ntile) {
    v16bf bk = frag_ld(s_k, 32, ntile * 16, 0);
    v8f z = {};
    s[ntile] = wmma_bf16(aq, bk, z);
  }

  // ---- row softmax over 64 (4 tiles x 16 lanes of the same 16-lane group) ----
#pragma unroll
  for (int r = 0; r < 8; ++r) {
    float v0 = s[0][r] * scaling, v1 = s[1][r] * scaling;
    float v2 = s[2][r] * scaling, v3 = s[3][r] * scaling;
    float mx = fmaxf(fmaxf(v0, v1), fmaxf(v2, v3));
#pragma unroll
    for (int d = 1; d < 16; d <<= 1) mx = fmaxf(mx, __shfl_xor(mx, d, 32));
    float e0 = __expf(v0 - mx), e1 = __expf(v1 - mx);
    float e2 = __expf(v2 - mx), e3 = __expf(v3 - mx);
    float sum = e0 + e1 + e2 + e3;
#pragma unroll
    for (int d = 1; d < 16; d <<= 1) sum += __shfl_xor(sum, d, 32);
    float inv = 1.f / sum;
    s[0][r] = e0 * inv; s[1][r] = e1 * inv; s[2][r] = e2 * inv; s[3][r] = e3 * inv;
  }

  // ---- transpose P through LDS, bf16 (C layout -> A-fragment layout) ----
  int mhalf = (lane & 16) ? 8 : 0;
  __bf16* pb = &s_p[wave][0];
#pragma unroll
  for (int r = 0; r < 8; ++r) {
    int m = mhalf + r;
#pragma unroll
    for (int ntile = 0; ntile < 4; ++ntile)
      pb[m * 64 + ntile * 16 + (lane & 15)] = (__bf16)s[ntile][r];
  }
  __syncthreads();

  // ---- XX = P @ V : K=64 (2 steps), N=32 (2 tiles) ----
  v8f acc0 = {}, acc1 = {};
#pragma unroll
  for (int k0 = 0; k0 < 64; k0 += 32) {
    v16bf ap  = frag_ld(pb, 64, 0, k0);
    v16bf bv0 = frag_ld(s_v, 64, 0, k0);
    v16bf bv1 = frag_ld(s_v, 64, 16, k0);
    acc0 = wmma_bf16(ap, bv0, acc0);
    acc1 = wmma_bf16(ap, bv1, acc1);
  }

  // ---- store bf16 into xx_cat: row = t*B + b (ld=512), col = colofs + nh*32 + n ----
#pragma unroll
  for (int r = 0; r < 8; ++r) {
    int t = trow + mhalf + r;
    long rowbase = ((long)t * B_DIM + b) * 512 + colofs + nh * HDIM;
    xx[rowbase + (lane & 15)]      = (__bf16)acc0[r];
    xx[rowbase + 16 + (lane & 15)] = (__bf16)acc1[r];
  }
}

// ---------- host launch ----------

static inline int gemm_blocks(int M, int N) {
  int tiles = ((M + 15) / 16) * (N / 64);
  return (tiles + 7) / 8;
}

extern "C" void kernel_launch(void* const* d_in, const int* in_sizes, int n_in,
                              void* d_out, int out_size, void* d_ws, size_t ws_size,
                              hipStream_t stream) {
  const float* query_row = (const float*)d_in[0];
  const float* query_col = (const float*)d_in[1];
  const float* key_row   = (const float*)d_in[2];
  const float* key_col   = (const float*)d_in[3];
  const float* value     = (const float*)d_in[4];
  const float* ipw       = (const float*)d_in[5];   // [5E, E]
  const float* ipb       = (const float*)d_in[6];   // [5E]
  const float* w_row     = (const float*)d_in[7];
  const float* b_row     = (const float*)d_in[8];
  const float* w_col     = (const float*)d_in[9];
  const float* b_col     = (const float*)d_in[10];
  const float* w_out     = (const float*)d_in[11];
  const float* b_out     = (const float*)d_in[12];
  const float* conv_w    = (const float*)d_in[13];
  const float* conv_b    = (const float*)d_in[14];
  float* out = (float*)d_out;

  char* wsb = (char*)d_ws;
  size_t off = 0;
  auto alloc = [&](size_t bytes) {
    void* p = wsb + off;
    off = (off + bytes + 255) & ~(size_t)255;
    return p;
  };

  // bf16 operand copies
  __bf16* qr_bf   = (__bf16*)alloc((size_t)T_DIM * B_DIM * E_DIM * 2);
  __bf16* qc_bf   = (__bf16*)alloc((size_t)T_DIM * B_DIM * E_DIM * 2);
  __bf16* ipw_bf  = (__bf16*)alloc((size_t)5 * E_DIM * E_DIM * 2);
  __bf16* wout_bf = (__bf16*)alloc((size_t)E_DIM * E_DIM * 2);
  __bf16* wcat_bf = (__bf16*)alloc((size_t)E_DIM * 512 * 2);
  float*  bias_sum = (float*)alloc(E_DIM * 4);
  // reduced means (bf16, GEMM inputs)
  __bf16* mean_kr = (__bf16*)alloc(65536 * 2);
  __bf16* mean_kc = (__bf16*)alloc(65536 * 2);
  __bf16* mean_vr = (__bf16*)alloc(65536 * 2);
  __bf16* mean_vc = (__bf16*)alloc(65536 * 2);
  __bf16* mean_v  = (__bf16*)alloc(1024 * 2);
  // projected means (f32, small)
  float* krp = (float*)alloc(65536 * 4);
  float* kcp = (float*)alloc(65536 * 4);
  float* vrp = (float*)alloc(65536 * 4);
  float* vcp = (float*)alloc(65536 * 4);
  float* vmp = (float*)alloc(1024 * 4);
  float* gate = (float*)alloc(1024 * 4);
  // per-head bf16 matrices for attention
  __bf16* krg = (__bf16*)alloc(65536 * 2);
  __bf16* kcg = (__bf16*)alloc(65536 * 2);
  __bf16* vrT = (__bf16*)alloc(65536 * 2);
  __bf16* vcT = (__bf16*)alloc(65536 * 2);
  // big intermediates (bf16)
  __bf16* qrp = (__bf16*)alloc((size_t)16384 * 256 * 2);
  __bf16* qcp = (__bf16*)alloc((size_t)16384 * 256 * 2);
  __bf16* xx_cat = (__bf16*)alloc((size_t)16384 * 512 * 2);
  __bf16* tmp_bf = (__bf16*)alloc((size_t)16384 * 256 * 2);

  const long HWE = (long)H_DIM * W_DIM * E_DIM;   // 1,048,576
  const long WE  = (long)W_DIM * E_DIM;           // 16,384

  // --- bf16 conversions of GEMM operands (one-time, bandwidth-bound) ---
  convert_kernel<<<4096, 256, 0, stream>>>(query_row, qr_bf, 1048576);
  convert_kernel<<<4096, 256, 0, stream>>>(query_col, qc_bf, 1048576);
  convert_kernel<<<320, 256, 0, stream>>>(ipw, ipw_bf, 81920);
  convert_kernel<<<64, 256, 0, stream>>>(w_out, wout_bf, 16384);
  pack_wcat_kernel<<<256, 256, 0, stream>>>(w_row, w_col, b_row, b_col,
                                            wcat_bf, bias_sum);

  // --- means (linearity lets us reduce before projecting) ---
  mean_kernel<<<256, 256, 0, stream>>>(key_row, mean_kr, (int)WE, H_DIM, WE, HWE,
                                       65536, 1.f / H_DIM);
  mean_kernel<<<256, 256, 0, stream>>>(key_col, mean_kc, E_DIM, W_DIM, E_DIM, WE,
                                       65536, 1.f / W_DIM);
  mean_kernel<<<256, 256, 0, stream>>>(value, mean_vr, (int)WE, H_DIM, WE, HWE,
                                       65536, 1.f / H_DIM);
  mean_kernel<<<256, 256, 0, stream>>>(value, mean_vc, E_DIM, W_DIM, E_DIM, WE,
                                       65536, 1.f / W_DIM);
  mean_kernel<<<4, 256, 0, stream>>>(value, mean_v, E_DIM, T_DIM, E_DIM, HWE,
                                     1024, 1.f / T_DIM);

  // --- projections ---
  const __bf16* Wq0 = ipw_bf + 0L * 65536; const float* bq0 = ipb + 0 * E_DIM;
  const __bf16* Wq1 = ipw_bf + 1L * 65536; const float* bq1 = ipb + 1 * E_DIM;
  const __bf16* Wk0 = ipw_bf + 2L * 65536; const float* bk0 = ipb + 2 * E_DIM;
  const __bf16* Wk1 = ipw_bf + 3L * 65536; const float* bk1 = ipb + 3 * E_DIM;
  const __bf16* Wv  = ipw_bf + 4L * 65536; const float* bv  = ipb + 4 * E_DIM;

  wmma_gemm_bf<<<gemm_blocks(16384, 256), 256, 0, stream>>>(
      qr_bf, Wq0, bq0, nullptr, qrp, 16384, 256, 256);
  wmma_gemm_bf<<<gemm_blocks(16384, 256), 256, 0, stream>>>(
      qc_bf, Wq1, bq1, nullptr, qcp, 16384, 256, 256);
  wmma_gemm_bf<<<gemm_blocks(256, 256), 256, 0, stream>>>(
      mean_kr, Wk0, bk0, krp, nullptr, 256, 256, 256);
  wmma_gemm_bf<<<gemm_blocks(256, 256), 256, 0, stream>>>(
      mean_kc, Wk1, bk1, kcp, nullptr, 256, 256, 256);
  wmma_gemm_bf<<<gemm_blocks(256, 256), 256, 0, stream>>>(
      mean_vr, Wv, bv, vrp, nullptr, 256, 256, 256);
  wmma_gemm_bf<<<gemm_blocks(256, 256), 256, 0, stream>>>(
      mean_vc, Wv, bv, vcp, nullptr, 256, 256, 256);
  wmma_gemm_bf<<<gemm_blocks(4, 256), 256, 0, stream>>>(
      mean_v, Wv, bv, vmp, nullptr, 4, 256, 256);

  // --- SE gate + head repack (gated K, transposed V), bf16 out ---
  gate_kernel<<<4, 256, 0, stream>>>(vmp, conv_w, conv_b, gate);
  repack_kernel<<<256, 256, 0, stream>>>(krp, kcp, vrp, vcp, gate,
                                         krg, kcg, vrT, vcT);

  // --- fused attention (row + col), outputs into concatenated [T*B, 512] ---
  const float scaling = 0.17677669529663689f;   // HD^-0.5
  dim3 agrid(T_DIM / 128, NHEADS_TOTAL);
  attn_kernel<<<agrid, 256, 0, stream>>>(qrp, krg, vrT, xx_cat, 0, scaling);
  attn_kernel<<<agrid, 256, 0, stream>>>(qcp, kcg, vcT, xx_cat, 256, scaling);

  // --- epilogue: tmp = xx_cat @ w_cat^T + (b_row+b_col)  (single K=512 GEMM) ---
  wmma_gemm_bf<<<gemm_blocks(16384, 256), 256, 0, stream>>>(
      xx_cat, wcat_bf, bias_sum, nullptr, tmp_bf, 16384, 256, 512);
  // --- out = tmp @ w_out^T + b_out ---
  wmma_gemm_bf<<<gemm_blocks(16384, 256), 256, 0, stream>>>(
      tmp_bf, wout_bf, b_out, out, nullptr, 16384, 256, 256);
}
